// LSTM_9148280340802
// MI455X (gfx1250) — matrix operand order repaired and checked by reading, not compiled, not run on hardware
//
#include <hip/hip_runtime.h>
#include <hip/hip_bf16.h>
#include <stddef.h>

#define H_DIM   1024
#define T_STEPS 256
#define B_DIM   64
#define L_LAYERS 4
#define NWG     64      // one workgroup per 16 hidden units (64*16 = 1024)
#define NTHR    128     // 4 waves of 32

typedef __bf16 bf16_t;
typedef bf16_t v16bf __attribute__((ext_vector_type(16)));
typedef float  v8f   __attribute__((ext_vector_type(8)));
typedef unsigned int v4u __attribute__((ext_vector_type(4)));

union Frag16 { v16bf bf; v4u u[2]; };

__device__ __forceinline__ unsigned short f2bf(float f) {
  unsigned int u = __float_as_uint(f);
  u += 0x7FFFu + ((u >> 16) & 1u);          // round-to-nearest-even
  return (unsigned short)(u >> 16);
}
__device__ __forceinline__ float sigmf(float x) {
  return 1.0f / (1.0f + __expf(-x));
}
__device__ __forceinline__ float tanh_fast(float x) {
  float cx = fminf(fmaxf(x, -15.0f), 15.0f);
  float e = __expf(-2.0f * cx);
  return (1.0f - e) / (1.0f + e);
}

__device__ __forceinline__ void sig_ctr(int* p) {
  __hip_atomic_fetch_add(p, 1, __ATOMIC_RELEASE, __HIP_MEMORY_SCOPE_AGENT);
}
__device__ __forceinline__ void wait_ctr(int* p, int tgt) {
  while (__hip_atomic_load(p, __ATOMIC_ACQUIRE, __HIP_MEMORY_SCOPE_AGENT) < tgt) {
    __builtin_amdgcn_s_sleep(2);
  }
}

// CDNA5 async copy: 16B global -> LDS, bypassing VGPRs, tracked by ASYNCcnt.
// lds_off = per-lane LDS byte address (HW adds wave LDS_BASE), gaddr = 64-bit VA.
__device__ __forceinline__ void async_g2lds_b128(unsigned lds_off, const void* gptr) {
  asm volatile("global_load_async_to_lds_b128 %0, %1, off"
               :
               : "v"(lds_off), "v"((unsigned long long)(size_t)gptr)
               : "memory");
}
__device__ __forceinline__ void wait_asynccnt0() {
  asm volatile("s_wait_asynccnt 0x0" ::: "memory");
}

// ---------------- prep kernels ----------------
__global__ void f32_to_bf16_kernel(const float* __restrict__ src,
                                   unsigned short* __restrict__ dst, int n) {
  int i = blockIdx.x * blockDim.x + threadIdx.x;
  if (i < n) dst[i] = f2bf(src[i]);
}
__global__ void bias_add_kernel(const float* __restrict__ a,
                                const float* __restrict__ b,
                                float* __restrict__ dst, int n) {
  int i = blockIdx.x * blockDim.x + threadIdx.x;
  if (i < n) dst[i] = a[i] + b[i];
}
__global__ void zero_i32_kernel(int* __restrict__ p, int n) {
  int i = blockIdx.x * blockDim.x + threadIdx.x;
  if (i < n) p[i] = 0;
}

// ---------------- persistent LSTM layer kernel ----------------
// Workgroup wg owns hidden units [wg*16, wg*16+16). Its 64 gate columns are
// the 4 gate blocks (i,f,g,o) x 16 units -> N-tile index == gate index.
// Wave w owns batch rows [w*16, w*16+16) (M-tile). K = 1024 (h) [+1024 (x)].
__global__ __launch_bounds__(NTHR) void lstm_layer_kernel(
    int layer,
    const float* __restrict__ x0,            // layer 0 raw input [B][T][2]
    const unsigned short* __restrict__ xs_bf, // prev-layer hs bf16 [B][T][H] (null for layer 0)
    const unsigned short* __restrict__ whh_bf,// this layer W_hh bf16 [4H][H]
    const unsigned short* __restrict__ wih_bf,// this layer W_ih bf16 [4H][H] (null for layer 0)
    const float* __restrict__ wih0,           // layer 0 W_ih f32 [4H][2]
    const float* __restrict__ bias,           // combined bias f32 [4H]
    unsigned short* __restrict__ hbuf,        // ping-pong h bf16 [2][B][H]
    unsigned short* __restrict__ hs_out,      // this layer hs bf16 [B][T][H] (null for last layer)
    float* __restrict__ hT_out,               // final h f32 [B][H] (last layer only)
    int* __restrict__ syw,                    // write-ready counters [T+1]
    int* __restrict__ syr)                    // read-done counters [T]
{
  extern __shared__ char smem[];
  unsigned short* whh_lds = (unsigned short*)smem;              // 64 cols x 1024 = 128 KB @ LDS off 0
  unsigned short* wih_lds = (unsigned short*)(smem + 131072);   // 64 cols x 1024 = 128 KB @ LDS off 131072
  float* c_lds    = (float*)(smem + 262144);                    // [64][16] cell state
  float* bias_lds = (float*)(smem + 266240);                    // 64
  float* wih0_lds = (float*)(smem + 266496);                    // [64][2]

  const int tid  = threadIdx.x;
  const int wave = tid >> 5;
  const int lane = tid & 31;
  const int wg   = blockIdx.x;
  const int u0   = wg * 16;

  // stage this WG's weight columns into LDS via CDNA5 async global->LDS DMA
  // (16B per lane per issue; dynamic LDS block starts at LDS byte offset 0)
  for (int idx = tid; idx < 64 * (H_DIM / 8); idx += NTHR) {
    int c  = idx >> 7;              // local column 0..63
    int k8 = (idx & 127) << 3;      // k offset (x8 elements)
    int grow = (c >> 4) * H_DIM + u0 + (c & 15);   // row in [4H]
    unsigned dst = (unsigned)(c * H_DIM + k8) * 2u;
    async_g2lds_b128(dst, whh_bf + (size_t)grow * H_DIM + k8);
    if (wih_bf)
      async_g2lds_b128(131072u + dst, wih_bf + (size_t)grow * H_DIM + k8);
  }
  for (int idx = tid; idx < 64; idx += NTHR) {
    int grow = (idx >> 4) * H_DIM + u0 + (idx & 15);
    bias_lds[idx] = bias[grow];
    if (wih0) {
      wih0_lds[idx * 2 + 0] = wih0[grow * 2 + 0];
      wih0_lds[idx * 2 + 1] = wih0[grow * 2 + 1];
    }
  }
  for (int idx = tid; idx < B_DIM * 16; idx += NTHR) c_lds[idx] = 0.0f;
  // zero our column slice of h ping buffer (input for t = 0)
  for (int idx = tid; idx < B_DIM * 16; idx += NTHR)
    hbuf[(idx >> 4) * H_DIM + u0 + (idx & 15)] = 0;

  wait_asynccnt0();                 // weight tiles landed in LDS
  __threadfence();
  __syncthreads();
  if (tid == 0) sig_ctr(&syw[0]);

  const int ncol  = lane & 15;        // N column within tile (= hidden unit j)
  const int khalf = lane >> 4;        // K-half select for A/B fragments
  const int mhalf = khalf * 8;        // M offset for C/D layout
  const int arow  = wave * 16 + ncol; // A-matrix row handled by this lane

  for (int t = 0; t < T_STEPS; ++t) {
    if (tid == 0) wait_ctr(&syw[t], NWG);   // h_{t-1} fully written
    __syncthreads();

    // init accumulators: bias (+ layer-0 input projection, K=2, via VALU)
    v8f acc[4];
    #pragma unroll
    for (int g = 0; g < 4; ++g) {
      float bb = bias_lds[g * 16 + ncol];
      #pragma unroll
      for (int r = 0; r < 8; ++r) acc[g][r] = bb;
    }
    if (layer == 0) {
      #pragma unroll
      for (int r = 0; r < 8; ++r) {
        int brow = wave * 16 + mhalf + r;
        const float2 xv = *(const float2*)(x0 + ((size_t)brow * T_STEPS + t) * 2);
        #pragma unroll
        for (int g = 0; g < 4; ++g) {
          int c = g * 16 + ncol;
          acc[g][r] += xv.x * wih0_lds[c * 2 + 0] + xv.y * wih0_lds[c * 2 + 1];
        }
      }
    }

    // recurrent GEMM: gates += h_{t-1} @ W_hh^T  (A from global, B from LDS)
    const unsigned short* hin = hbuf + (size_t)(t & 1) * (B_DIM * H_DIM);
    const unsigned short* ar  = hin + (size_t)arow * H_DIM;
    #pragma unroll 4
    for (int k0 = 0; k0 < H_DIM; k0 += 32) {
      Frag16 A;
      A.u[0] = *(const v4u*)(ar + k0 + khalf * 8);
      A.u[1] = *(const v4u*)(ar + k0 + 16 + khalf * 8);
      #pragma unroll
      for (int g = 0; g < 4; ++g) {
        const unsigned short* wp = whh_lds + (g * 16 + ncol) * H_DIM + k0 + khalf * 8;
        Frag16 Bm;
        Bm.u[0] = *(const v4u*)(wp);
        Bm.u[1] = *(const v4u*)(wp + 16);
        acc[g] = __builtin_amdgcn_wmma_f32_16x16x32_bf16(
            false, A.bf, false, Bm.bf, (short)0, acc[g], false, false);
      }
    }

    // input GEMM (layers >= 1): gates += x_t @ W_ih^T
    if (xs_bf) {
      const unsigned short* xr = xs_bf + ((size_t)arow * T_STEPS + t) * H_DIM;
      #pragma unroll 4
      for (int k0 = 0; k0 < H_DIM; k0 += 32) {
        Frag16 A;
        A.u[0] = *(const v4u*)(xr + k0 + khalf * 8);
        A.u[1] = *(const v4u*)(xr + k0 + 16 + khalf * 8);
        #pragma unroll
        for (int g = 0; g < 4; ++g) {
          const unsigned short* wp = wih_lds + (g * 16 + ncol) * H_DIM + k0 + khalf * 8;
          Frag16 Bm;
          Bm.u[0] = *(const v4u*)(wp);
          Bm.u[1] = *(const v4u*)(wp + 16);
          acc[g] = __builtin_amdgcn_wmma_f32_16x16x32_bf16(
              false, A.bf, false, Bm.bf, (short)0, acc[g], false, false);
        }
      }
      if (t + 1 < T_STEPS) __builtin_prefetch(xr + H_DIM, 0, 0); // next x_t row
    }

    __syncthreads();                 // all waves finished reading h_in
    if (tid == 0) {
      sig_ctr(&syr[t]);                              // done reading buf[t&1]
      if (t > 0) wait_ctr(&syr[t - 1], NWG);         // safe to overwrite buf[(t+1)&1]
    }
    __syncthreads();

    // elementwise LSTM cell update; all 4 gate values for (b,j) are local
    unsigned short* hout = hbuf + (size_t)((t + 1) & 1) * (B_DIM * H_DIM);
    #pragma unroll
    for (int r = 0; r < 8; ++r) {
      int brow = wave * 16 + mhalf + r;
      float gi = sigmf(acc[0][r]);
      float gf = sigmf(acc[1][r]);
      float gg = tanh_fast(acc[2][r]);
      float go = sigmf(acc[3][r]);
      int ci = brow * 16 + ncol;
      float cv = gf * c_lds[ci] + gi * gg;
      c_lds[ci] = cv;
      float hv = go * tanh_fast(cv);
      unsigned short hb = f2bf(hv);
      hout[brow * H_DIM + u0 + ncol] = hb;
      if (hs_out)
        hs_out[((size_t)brow * T_STEPS + t) * H_DIM + u0 + ncol] = hb;
      if (hT_out != nullptr && t == T_STEPS - 1)
        hT_out[brow * H_DIM + u0 + ncol] = hv;
    }
    __threadfence();
    __syncthreads();
    if (tid == 0) sig_ctr(&syw[t + 1]);   // h_t published
  }
}

// ---------------- final FC: out[b] = dot(hT[b], fc_w) + fc_b ----------------
__global__ __launch_bounds__(128) void fc_kernel(const float* __restrict__ hT,
                                                 const float* __restrict__ w,
                                                 const float* __restrict__ bterm,
                                                 float* __restrict__ out) {
  __shared__ float red[128];
  int b = blockIdx.x;
  float s = 0.0f;
  for (int k = threadIdx.x; k < H_DIM; k += 128)
    s += hT[(size_t)b * H_DIM + k] * w[k];
  red[threadIdx.x] = s;
  __syncthreads();
  for (int off = 64; off > 0; off >>= 1) {
    if (threadIdx.x < off) red[threadIdx.x] += red[threadIdx.x + off];
    __syncthreads();
  }
  if (threadIdx.x == 0) out[b] = red[0] + bterm[0];
}

// ---------------- host ----------------
extern "C" void kernel_launch(void* const* d_in, const int* in_sizes, int n_in,
                              void* d_out, int out_size, void* d_ws, size_t ws_size,
                              hipStream_t stream) {
  (void)in_sizes; (void)n_in; (void)out_size; (void)ws_size;
  const float* x     = (const float*)d_in[0];  // [64][256][2]
  const float* w_ih0 = (const float*)d_in[1];  // [4096][2]
  const float* w_ihr = (const float*)d_in[2];  // [3][4096][1024]
  const float* w_hh  = (const float*)d_in[3];  // [4][4096][1024]
  const float* b_ih  = (const float*)d_in[4];  // [4][4096]
  const float* b_hh  = (const float*)d_in[5];  // [4][4096]
  const float* fc_w  = (const float*)d_in[6];  // [1024]
  const float* fc_b  = (const float*)d_in[7];  // [1]
  float* out = (float*)d_out;                  // [64][1] f32

  char* ws = (char*)d_ws;
  // workspace layout (~121 MB)
  const size_t OFF_WHH  = 0;                         // 4*4096*1024 bf16 = 32 MB
  const size_t OFF_WIH  = 33554432;                  // 3*4096*1024 bf16 = 24 MB
  const size_t OFF_BIAS = 58720256;                  // 4*4096 f32
  const size_t OFF_HSA  = 58785792;                  // [64][256][1024] bf16 = 32 MB
  const size_t OFF_HSB  = 92340224;                  // 32 MB
  const size_t OFF_HBUF = 125894656;                 // [2][64][1024] bf16
  const size_t OFF_HT   = 126156800;                 // [64][1024] f32
  const size_t OFF_SYNC = 126418944;                 // counters

  unsigned short* whh_bf = (unsigned short*)(ws + OFF_WHH);
  unsigned short* wih_bf = (unsigned short*)(ws + OFF_WIH);
  float*          bias_w = (float*)(ws + OFF_BIAS);
  unsigned short* hs_a   = (unsigned short*)(ws + OFF_HSA);
  unsigned short* hs_b   = (unsigned short*)(ws + OFF_HSB);
  unsigned short* hbuf   = (unsigned short*)(ws + OFF_HBUF);
  float*          hT     = (float*)(ws + OFF_HT);
  int*            syw    = (int*)(ws + OFF_SYNC);                    // [4][257]
  int*            syr    = syw + L_LAYERS * (T_STEPS + 1);           // [4][256]

  // prep: bf16 weight conversion, combined bias, zeroed barrier counters
  int n_whh = L_LAYERS * 4 * H_DIM * H_DIM;          // 16,777,216
  int n_wih = (L_LAYERS - 1) * 4 * H_DIM * H_DIM;    // 12,582,912
  int n_b   = L_LAYERS * 4 * H_DIM;                  // 16,384
  int n_sy  = L_LAYERS * (T_STEPS + 1) + L_LAYERS * T_STEPS;
  f32_to_bf16_kernel<<<(n_whh + 255) / 256, 256, 0, stream>>>(w_hh, whh_bf, n_whh);
  f32_to_bf16_kernel<<<(n_wih + 255) / 256, 256, 0, stream>>>(w_ihr, wih_bf, n_wih);
  bias_add_kernel<<<(n_b + 255) / 256, 256, 0, stream>>>(b_ih, b_hh, bias_w, n_b);
  zero_i32_kernel<<<(n_sy + 255) / 256, 256, 0, stream>>>(syw, n_sy);

  const int SMEM_BYTES = 267008;  // 2x128KB weight slices + c-state + bias + wih0
  hipFuncSetAttribute((const void*)lstm_layer_kernel,
                      hipFuncAttributeMaxDynamicSharedMemorySize, SMEM_BYTES);

  unsigned short* hs_io[5] = {nullptr, hs_a, hs_b, hs_a, hs_b};
  for (int l = 0; l < L_LAYERS; ++l) {
    const unsigned short* xs_in = hs_io[l];                 // null for layer 0
    unsigned short* hs_out = (l == L_LAYERS - 1) ? nullptr : hs_io[l + 1];
    lstm_layer_kernel<<<NWG, NTHR, SMEM_BYTES, stream>>>(
        l,
        (l == 0) ? x : nullptr,
        xs_in,
        whh_bf + (size_t)l * 4 * H_DIM * H_DIM,
        (l == 0) ? nullptr : (wih_bf + (size_t)(l - 1) * 4 * H_DIM * H_DIM),
        (l == 0) ? w_ih0 : nullptr,
        bias_w + (size_t)l * 4 * H_DIM,
        hbuf,
        hs_out,
        (l == L_LAYERS - 1) ? hT : nullptr,
        syw + l * (T_STEPS + 1),
        syr + l * T_STEPS);
  }

  fc_kernel<<<B_DIM, 128, 0, stream>>>(hT, fc_w, fc_b, out);
}
